// lstmcell_45578192945385
// MI455X (gfx1250) — matrix operand order repaired and checked by reading, not compiled, not run on hardware
//
#include <hip/hip_runtime.h>
#include <hip/hip_bf16.h>

// LSTM cell fused kernel for MI455X (gfx1250, wave32, WMMA).
// Compute-bound GEMM (AI ~780 FLOP/B): bf16 WMMA w/ fp32 accumulate,
// double-buffered LDS (1 barrier / 64-K stage, 16 WMMAs / barrier),
// async global->LDS A-tile loads (GLOBAL_LOAD_ASYNC_TO_LDS_B128, ASYNCcnt).
// Requires 80 MB of d_ws for bf16 shadows of x/h/Wx/Wh.

typedef __attribute__((ext_vector_type(16))) __bf16 v16bf;
typedef __attribute__((ext_vector_type(8)))  __bf16 v8bf;
typedef __attribute__((ext_vector_type(4)))  __bf16 v4bf;
typedef __attribute__((ext_vector_type(8)))  float  v8f;
typedef __attribute__((ext_vector_type(4)))  int    v4i;

#define D_IN   1024
#define D_OUT  1024
#define BATCH  16384

#define BLOCK_M 128    // batch rows per workgroup
#define BLOCK_N 32     // output cols per workgroup (x 4 gates)
#define KOUTER  64     // K depth per LDS stage (2 WMMA K-steps)
#define NKT     ((2 * D_IN) / KOUTER)   // 32 stages (x*Wx then h*Wh)
#define A_STRIDE 72    // 64 + 8 halfword pad (144B rows: 16B-aligned, conflict-breaking)
#define B_STRIDE 72

#if defined(__has_builtin)
#if __has_builtin(__builtin_amdgcn_global_load_async_to_lds_b128) && \
    __has_builtin(__builtin_amdgcn_s_wait_asynccnt)
#define USE_ASYNC_A 1
#endif
#endif

#ifdef USE_ASYNC_A
// Builtin signature (from compiler diagnostic): arg0 = v4i in AS(1) (global),
// arg1 = v4i in AS(3) (LDS), then two immediate ints (offset, cpol).
typedef __attribute__((address_space(1))) v4i* g_v4i_ptr;
typedef __attribute__((address_space(3))) v4i* l_v4i_ptr;

__device__ __forceinline__ g_v4i_ptr as_g(const void* p) {
    return (g_v4i_ptr)(unsigned long long)p;
}
__device__ __forceinline__ l_v4i_ptr as_l(const void* p) {
    // LDS generic addresses carry the LDS byte offset in the low 32 bits.
    return (l_v4i_ptr)(unsigned int)(unsigned long long)p;
}
#endif

__global__ void f32_to_bf16_kernel(const float* __restrict__ src,
                                   __bf16* __restrict__ dst, int n) {
    int i = (blockIdx.x * blockDim.x + threadIdx.x) * 4;
    if (i < n) {
        float4 v = *(const float4*)(src + i);
        v4bf o;
        o[0] = (__bf16)v.x; o[1] = (__bf16)v.y;
        o[2] = (__bf16)v.z; o[3] = (__bf16)v.w;
        *(v4bf*)(dst + i) = o;
    }
}

__global__ __launch_bounds__(256)
void lstm_wmma_kernel(const __bf16* __restrict__ xbf,
                      const __bf16* __restrict__ hbf,
                      const __bf16* __restrict__ wxbf,   // [4, D_IN, D_OUT]
                      const __bf16* __restrict__ whbf,   // [4, D_IN, D_OUT]
                      const float*  __restrict__ bx,     // [4, D_OUT]
                      const float*  __restrict__ bh,     // [4, D_OUT]
                      const float*  __restrict__ cprev,  // [BATCH, D_OUT]
                      float* __restrict__ out,           // [BATCH, D_OUT]
                      float* __restrict__ cellout)       // [BATCH, D_OUT]
{
    __shared__ __align__(16) __bf16 Ash[2][BLOCK_M][A_STRIDE];      // [buf][row][k]
    __shared__ __align__(16) __bf16 Bsh[2][4][BLOCK_N][B_STRIDE];   // [buf][gate][n][k] (transposed)

    const int tid    = threadIdx.x;
    const int wave   = tid >> 5;
    const int lane   = tid & 31;
    const int wave_m = wave & 3;        // 4 wave-rows  -> 32 batch rows each
    const int wave_n = wave >> 2;       // 2 wave-cols  -> 16 out cols each

    const int blockCol = blockIdx.x * BLOCK_N;
    const int blockRow = blockIdx.y * BLOCK_M;

    const v8f vzero = {0.f, 0.f, 0.f, 0.f, 0.f, 0.f, 0.f, 0.f};
    v8f acc[4][2];                      // [gate][m-subtile]
    #pragma unroll
    for (int g = 0; g < 4; ++g) { acc[g][0] = vzero; acc[g][1] = vzero; }

    // ---- cooperative loader assignments (256 threads) ----
    // B: 4 gate tiles of 64(K) x 32(N); 64 threads per gate, one K-row each.
    const int b_gate = tid >> 6;
    const int b_krow = tid & 63;

    // ---- per-lane WMMA fragment coordinates (ISA §7.12.2, 16-bit layouts) ----
    const int am = lane & 15;
    const int ak = (lane >> 4) * 8;     // lanes 0-15: K 0-7,16-23 ; 16-31: K 8-15,24-31
    const int bn = lane & 15;
    const int bk = (lane >> 4) * 8;     // lanes 0-15: K 0-7,16-23 ; 16-31: K 8-15,24-31

    // =================== prologue: stage tile 0 ===================
    v8bf breg[4];
    {
        const __bf16* wp = wxbf + ((size_t)b_gate * D_IN + b_krow) * D_OUT + blockCol;
        breg[0] = *(const v8bf*)(wp);
        breg[1] = *(const v8bf*)(wp + 8);
        breg[2] = *(const v8bf*)(wp + 16);
        breg[3] = *(const v8bf*)(wp + 24);
    }
#ifdef USE_ASYNC_A
    #pragma unroll
    for (int it = 0; it < 4; ++it) {    // A tile 0: 128 rows x 128 B -> 1024 b128 chunks
        const int c = it * 256 + tid, row = c >> 3, q = c & 7;
        const __bf16* ap = xbf + (size_t)(blockRow + row) * D_IN + q * 8;
        __builtin_amdgcn_global_load_async_to_lds_b128(
            as_g(ap), as_l(&Ash[0][row][q * 8]), 0, 0);
    }
#else
    v8bf areg[4];
    #pragma unroll
    for (int it = 0; it < 4; ++it) {
        const int c = it * 256 + tid, row = c >> 3, q = c & 7;
        areg[it] = *(const v8bf*)(xbf + (size_t)(blockRow + row) * D_IN + q * 8);
    }
#endif

    // =================== main loop: 32 stages of K=64 ===================
    #pragma unroll 1
    for (int kt = 0; kt < NKT; ++kt) {
        const int cur = kt & 1;
        const int nxt = cur ^ 1;

        // ---- commit staged B (transpose into [n][k]) ----
        #pragma unroll
        for (int v = 0; v < 4; ++v)
            #pragma unroll
            for (int j = 0; j < 8; ++j)
                Bsh[cur][b_gate][v * 8 + j][b_krow] = breg[v][j];
#ifndef USE_ASYNC_A
        #pragma unroll
        for (int it = 0; it < 4; ++it) {
            const int c = it * 256 + tid, row = c >> 3, q = c & 7;
            *(v8bf*)&Ash[cur][row][q * 8] = areg[it];
        }
#endif

        // ---- prefetch next stage's B (and A regs in fallback) ----
        const int  ktn = kt + 1;
        const bool fh  = ktn < (NKT / 2);
        const __bf16* An  = fh ? xbf  : hbf;
        const __bf16* Wn  = fh ? wxbf : whbf;
        const int     k0n = (fh ? ktn : ktn - (NKT / 2)) * KOUTER;
        if (ktn < NKT) {
            const __bf16* wp = Wn + ((size_t)b_gate * D_IN + (size_t)(k0n + b_krow)) * D_OUT
                                  + blockCol;
            breg[0] = *(const v8bf*)(wp);
            breg[1] = *(const v8bf*)(wp + 8);
            breg[2] = *(const v8bf*)(wp + 16);
            breg[3] = *(const v8bf*)(wp + 24);
#ifndef USE_ASYNC_A
            #pragma unroll
            for (int it = 0; it < 4; ++it) {
                const int c = it * 256 + tid, row = c >> 3, q = c & 7;
                areg[it] = *(const v8bf*)(An + (size_t)(blockRow + row) * D_IN + k0n + q * 8);
            }
#endif
        }

#ifdef USE_ASYNC_A
        __builtin_amdgcn_s_wait_asynccnt(0);   // A(kt) landed in Ash[cur]
#endif
        __syncthreads();

#ifdef USE_ASYNC_A
        // Issue A(kt+1) -> Ash[nxt] AFTER the barrier: all waves have finished
        // reading Ash[nxt] (stage kt-1) before any wave passes this barrier.
        if (ktn < NKT) {
            #pragma unroll
            for (int it = 0; it < 4; ++it) {
                const int c = it * 256 + tid, row = c >> 3, q = c & 7;
                const __bf16* ap = An + (size_t)(blockRow + row) * D_IN + k0n + q * 8;
                __builtin_amdgcn_global_load_async_to_lds_b128(
                    as_g(ap), as_l(&Ash[nxt][row][q * 8]), 0, 0);
            }
        }
#endif

        // ---- compute: 2 K-steps x (4 gates x 2 m-subtiles) = 16 WMMAs ----
        #pragma unroll
        for (int ks = 0; ks < 2; ++ks) {
            union { v16bf v; v8bf h[2]; } afr[2], bfr[4];
            #pragma unroll
            for (int mt = 0; mt < 2; ++mt) {
                const __bf16* base = &Ash[cur][wave_m * 32 + mt * 16 + am][ks * 32 + ak];
                afr[mt].h[0] = *(const v8bf*)(base);
                afr[mt].h[1] = *(const v8bf*)(base + 16);
            }
            #pragma unroll
            for (int g = 0; g < 4; ++g) {
                const __bf16* base = &Bsh[cur][g][wave_n * 16 + bn][ks * 32 + bk];
                bfr[g].h[0] = *(const v8bf*)(base);
                bfr[g].h[1] = *(const v8bf*)(base + 16);
            }
            #pragma unroll
            for (int g = 0; g < 4; ++g)
                #pragma unroll
                for (int mt = 0; mt < 2; ++mt)
                    acc[g][mt] = __builtin_amdgcn_wmma_f32_16x16x32_bf16(
                        false, afr[mt].v, false, bfr[g].v,
                        (short)0, acc[g][mt], false, false);
        }
    }

    // =================== fused LSTM epilogue ===================
    // C/D layout: element j of v8f acc -> m = j + (lane>>4)*8, n = lane&15.
    const int n_lane = lane & 15;
    const int m_half = (lane >> 4) * 8;
    const int col = blockCol + wave_n * 16 + n_lane;
    float bias[4];
    #pragma unroll
    for (int g = 0; g < 4; ++g)
        bias[g] = bx[g * D_OUT + col] + bh[g * D_OUT + col];

    #pragma unroll
    for (int mt = 0; mt < 2; ++mt) {
        #pragma unroll
        for (int j = 0; j < 8; ++j) {
            const int row = blockRow + wave_m * 32 + mt * 16 + m_half + j;
            const float gi = acc[0][mt][j] + bias[0];
            const float gf = acc[1][mt][j] + bias[1];
            const float gc = acc[2][mt][j] + bias[2];
            const float go = acc[3][mt][j] + bias[3];
            const float i_g = 1.0f / (1.0f + __expf(-gi));
            const float f_g = 1.0f / (1.0f + __expf(-gf));
            const float c_g = tanhf(gc);
            const float o_g = 1.0f / (1.0f + __expf(-go));
            const size_t idx = (size_t)row * D_OUT + col;
            const float c_new = f_g * cprev[idx] + i_g * c_g;
            out[idx]     = o_g * tanhf(c_new);
            cellout[idx] = c_new;
        }
    }
}

extern "C" void kernel_launch(void* const* d_in, const int* in_sizes, int n_in,
                              void* d_out, int out_size, void* d_ws, size_t ws_size,
                              hipStream_t stream) {
    const float* x  = (const float*)d_in[0];   // [BATCH, D_IN]
    const float* h  = (const float*)d_in[1];   // [BATCH, D_IN]
    const float* c  = (const float*)d_in[2];   // [BATCH, D_OUT]
    const float* Wx = (const float*)d_in[3];   // [4, D_IN, D_OUT]
    const float* bx = (const float*)d_in[4];   // [4, D_OUT]
    const float* Wh = (const float*)d_in[5];   // [4, D_IN, D_OUT]
    const float* bh = (const float*)d_in[6];   // [4, D_OUT]
    float* out = (float*)d_out;                // [output | current_cell]

    // Workspace: bf16 shadows, 32 + 32 + 8 + 8 = 80 MB
    __bf16* xbf  = (__bf16*)d_ws;
    __bf16* hbf  = xbf  + (size_t)BATCH * D_IN;
    __bf16* wxbf = hbf  + (size_t)BATCH * D_IN;
    __bf16* whbf = wxbf + (size_t)4 * D_IN * D_OUT;

    const int nx = BATCH * D_IN;
    const int nw = 4 * D_IN * D_OUT;
    f32_to_bf16_kernel<<<(nx / 4 + 255) / 256, 256, 0, stream>>>(x,  xbf,  nx);
    f32_to_bf16_kernel<<<(nx / 4 + 255) / 256, 256, 0, stream>>>(h,  hbf,  nx);
    f32_to_bf16_kernel<<<(nw / 4 + 255) / 256, 256, 0, stream>>>(Wx, wxbf, nw);
    f32_to_bf16_kernel<<<(nw / 4 + 255) / 256, 256, 0, stream>>>(Wh, whbf, nw);

    dim3 grid(D_OUT / BLOCK_N, BATCH / BLOCK_M);   // 32 x 128 workgroups
    lstm_wmma_kernel<<<grid, 256, 0, stream>>>(
        xbf, hbf, wxbf, whbf, bx, bh, c,
        out, out + (size_t)BATCH * D_OUT);
}
